// BlockIndexNet_85435489452607
// MI455X (gfx1250) — matrix operand order repaired and checked by reading, not compiled, not run on hardware
//
#include <hip/hip_runtime.h>
#include <cstdint>
#include <cstddef>

// ---------------------------------------------------------------------------
// Grouped 3-layer MLP (8 species x 25000 atoms, 256->256->256->128, SiLU),
// fully fused per 128-row tile. bf16 WMMA with f32 accumulate.
// ---------------------------------------------------------------------------

typedef __attribute__((ext_vector_type(16))) __bf16 v16bf;
typedef __attribute__((ext_vector_type(8)))  __bf16 v8bf;
typedef __attribute__((ext_vector_type(8)))  float  v8f;
typedef __attribute__((ext_vector_type(8)))  unsigned short us8;

#define NSPEC 8
#define BLK   25000
#define DIN   256
#define NH1   256
#define NH2   256
#define DOUT  128
#define TM    128                 // rows per workgroup tile
#define STR   (256 + 8)           // padded LDS row stride (bf16 elems) to spread banks
#define TILES 196                 // ceil(25000/128)

__device__ __forceinline__ unsigned short f2bf_u16(float f) {
  union { float f; uint32_t u; } c; c.f = f;
  uint32_t r = c.u + 0x7FFFu + ((c.u >> 16) & 1u);      // round-to-nearest-even
  if ((c.u & 0x7F800000u) == 0x7F800000u) r = c.u;      // inf/nan passthrough
  return (unsigned short)(r >> 16);
}

__device__ __forceinline__ v16bf cat8(v8bf lo, v8bf hi) {
  return __builtin_shufflevector(lo, hi, 0,1,2,3,4,5,6,7,8,9,10,11,12,13,14,15);
}

__device__ __forceinline__ v8f zero8() {
  v8f z = {0.f,0.f,0.f,0.f,0.f,0.f,0.f,0.f};
  return z;
}

// A operand: 16x32 bf16 tile from LDS (row-major, stride STR).
// ISA layout: lanes 0-15 row M=lane, K {kb..kb+7, kb+16..kb+23} with kb=0;
// lanes 16-31 same rows, kb=8.
__device__ __forceinline__ v16bf load_a(const unsigned short* lds, int lane, int mt, int kt) {
  const int row = mt * 16 + (lane & 15);
  const int kb  = kt * 32 + ((lane >> 4) << 3);
  const unsigned short* p = lds + row * STR + kb;
  us8 lo = *(const us8*)(p);        // K kb .. kb+7
  us8 hi = *(const us8*)(p + 16);   // K kb+16 .. kb+23
  return cat8(__builtin_bit_cast(v8bf, lo), __builtin_bit_cast(v8bf, hi));
}

// B operand: 32x16 bf16 tile from N-major (pre-transposed) weights in global.
// Layout: lanes 0-15 column n, K {kb..kb+7, kb+16..kb+23} with kb=kt*32; lanes 16-31 kb=kt*32+8.
__device__ __forceinline__ v16bf load_b(const __bf16* wt, int lane, int n0, int kt, int K) {
  const int n  = n0 + (lane & 15);
  const int kb = kt * 32 + ((lane >> 4) << 3);
  const __bf16* p = wt + (size_t)n * K + kb;
  v8bf lo = *(const v8bf*)(p);
  v8bf hi = *(const v8bf*)(p + 16);
  return cat8(lo, hi);
}

// One hidden layer for this wave's 32-column strip: D = silu(A*W + b) -> LDS bf16.
__device__ __forceinline__ void hidden_layer(const unsigned short* __restrict__ ldsIn,
                                             unsigned short* __restrict__ ldsOut,
                                             const __bf16* __restrict__ wt,
                                             const float* __restrict__ bias,
                                             int lane, int wave) {
  const int n0 = wave * 32;
  v8f acc0[8], acc1[8];
#pragma unroll
  for (int mt = 0; mt < 8; ++mt) { acc0[mt] = zero8(); acc1[mt] = zero8(); }

#pragma unroll
  for (int kt = 0; kt < 8; ++kt) {
    v16bf b0 = load_b(wt, lane, n0,      kt, 256);
    v16bf b1 = load_b(wt, lane, n0 + 16, kt, 256);
#pragma unroll
    for (int mt = 0; mt < 8; ++mt) {
      v16bf a = load_a(ldsIn, lane, mt, kt);
      acc0[mt] = __builtin_amdgcn_wmma_f32_16x16x32_bf16(false, a, false, b0,
                                                         (short)0, acc0[mt], false, false);
      acc1[mt] = __builtin_amdgcn_wmma_f32_16x16x32_bf16(false, a, false, b1,
                                                         (short)0, acc1[mt], false, false);
    }
  }

  // Epilogue: bias + SiLU, store bf16 into LDS (becomes next layer's A).
  const int nl     = lane & 15;
  const int rowAdd = (lane >> 4) << 3;   // C/D layout: hi lanes hold M+8
  const float bv0 = bias[n0 + nl];
  const float bv1 = bias[n0 + 16 + nl];
#pragma unroll
  for (int mt = 0; mt < 8; ++mt) {
#pragma unroll
    for (int r = 0; r < 8; ++r) {
      const int row = mt * 16 + r + rowAdd;
      float v0 = acc0[mt][r] + bv0;
      v0 = v0 / (1.f + __expf(-v0));
      ldsOut[row * STR + n0 + nl] = f2bf_u16(v0);
      float v1 = acc1[mt][r] + bv1;
      v1 = v1 / (1.f + __expf(-v1));
      ldsOut[row * STR + n0 + 16 + nl] = f2bf_u16(v1);
    }
  }
}

__global__ void __launch_bounds__(256)
fused_mlp(const float* __restrict__ x, const int* __restrict__ bidx,
          const float* __restrict__ b1, const float* __restrict__ b2,
          const float* __restrict__ b3,
          const __bf16* __restrict__ w1t, const __bf16* __restrict__ w2t,
          const __bf16* __restrict__ w3t, float* __restrict__ out) {
  extern __shared__ unsigned short smem[];
  unsigned short* lds0 = smem;                  // x tile, then h2
  unsigned short* lds1 = smem + TM * STR;       // h1
  int* atomIdx = (int*)(smem + 2 * TM * STR);   // gathered atom ids (128)

  const int s    = blockIdx.x / TILES;
  const int tile = blockIdx.x % TILES;
  const int row0 = tile * TM;
  const int lane = threadIdx.x & 31;
  const int wave = threadIdx.x >> 5;

  // ---- Stage x tile: gather via block_index, fp32 -> bf16 -> LDS ----
  {
    const int r    = threadIdx.x >> 1;          // 0..127
    const int c0   = (threadIdx.x & 1) * 128;
    const int grow = row0 + r;
    const int atom = (grow < BLK) ? bidx[s * BLK + grow] : -1;
    if ((threadIdx.x & 1) == 0) atomIdx[r] = atom;
    const float* src = x + (size_t)(atom < 0 ? 0 : atom) * DIN + c0;
    unsigned short* dst = lds0 + r * STR + c0;
#pragma unroll
    for (int c = 0; c < 128; c += 8) {
      float4 f0, f1;
      if (atom >= 0) {
        f0 = *(const float4*)(src + c);
        f1 = *(const float4*)(src + c + 4);
      } else {
        f0 = make_float4(0.f, 0.f, 0.f, 0.f);
        f1 = f0;
      }
      ushort4 u0, u1;
      u0.x = f2bf_u16(f0.x); u0.y = f2bf_u16(f0.y);
      u0.z = f2bf_u16(f0.z); u0.w = f2bf_u16(f0.w);
      u1.x = f2bf_u16(f1.x); u1.y = f2bf_u16(f1.y);
      u1.z = f2bf_u16(f1.z); u1.w = f2bf_u16(f1.w);
      *(ushort4*)(dst + c)     = u0;
      *(ushort4*)(dst + c + 4) = u1;
    }
  }
  __syncthreads();

  // ---- Layer 1: h1 = silu(x W1 + b1) ----
  hidden_layer(lds0, lds1, w1t + (size_t)s * NH1 * DIN, b1 + s * NH1, lane, wave);
  __syncthreads();

  // ---- Layer 2: h2 = silu(h1 W2 + b2) (reuse lds0) ----
  hidden_layer(lds1, lds0, w2t + (size_t)s * NH2 * NH1, b2 + s * NH2, lane, wave);
  __syncthreads();

  // ---- Layer 3: o = h2 W3 + b3, scatter f32 to global ----
  {
    const __bf16* wt = w3t + (size_t)s * DOUT * NH2;
    const int n0 = wave * 16;                   // 8 waves x 16 cols = 128
    v8f acc[8];
#pragma unroll
    for (int mt = 0; mt < 8; ++mt) acc[mt] = zero8();
#pragma unroll
    for (int kt = 0; kt < 8; ++kt) {
      v16bf b = load_b(wt, lane, n0, kt, 256);
#pragma unroll
      for (int mt = 0; mt < 8; ++mt) {
        v16bf a = load_a(lds0, lane, mt, kt);
        acc[mt] = __builtin_amdgcn_wmma_f32_16x16x32_bf16(false, a, false, b,
                                                          (short)0, acc[mt], false, false);
      }
    }
    const int nl     = lane & 15;
    const int rowAdd = (lane >> 4) << 3;
    const float bv = b3[s * DOUT + n0 + nl];
#pragma unroll
    for (int mt = 0; mt < 8; ++mt) {
#pragma unroll
      for (int r = 0; r < 8; ++r) {
        const int row  = mt * 16 + r + rowAdd;
        const int atom = atomIdx[row];
        if (atom >= 0)
          out[(size_t)atom * DOUT + n0 + nl] = acc[mt][r] + bv;
      }
    }
  }
}

// ---------------------------------------------------------------------------
// Weight prep: fp32 K-major -> bf16 N-major ("transposed") in workspace.
// ---------------------------------------------------------------------------
__global__ void prep_weights(const float* __restrict__ W1, const float* __restrict__ W2,
                             const float* __restrict__ W3, unsigned short* __restrict__ ws) {
  const int i  = blockIdx.x * blockDim.x + threadIdx.x;
  const int n1 = NSPEC * NH1 * DIN;
  const int n2 = NSPEC * NH2 * NH1;
  const int n3 = NSPEC * DOUT * NH2;
  if (i < n1) {                                   // W1t[s][h][d] = W1[s][d][h]
    int s = i / (NH1 * DIN); int r = i % (NH1 * DIN);
    int h = r / DIN;         int d = r % DIN;
    ws[i] = f2bf_u16(W1[((size_t)s * DIN + d) * NH1 + h]);
  } else if (i < n1 + n2) {                       // W2t[s][k][h] = W2[s][h][k]
    int j = i - n1;
    int s = j / (NH2 * NH1); int r = j % (NH2 * NH1);
    int k = r / NH1;         int h = r % NH1;
    ws[i] = f2bf_u16(W2[((size_t)s * NH1 + h) * NH2 + k]);
  } else if (i < n1 + n2 + n3) {                  // W3t[s][o][k] = W3[s][k][o]
    int j = i - (n1 + n2);
    int s = j / (DOUT * NH2); int r = j % (DOUT * NH2);
    int o = r / NH2;          int k = r % NH2;
    ws[i] = f2bf_u16(W3[((size_t)s * NH2 + k) * DOUT + o]);
  }
}

extern "C" void kernel_launch(void* const* d_in, const int* in_sizes, int n_in,
                              void* d_out, int out_size, void* d_ws, size_t ws_size,
                              hipStream_t stream) {
  (void)in_sizes; (void)n_in; (void)out_size; (void)ws_size;
  const float* emb = (const float*)d_in[0];
  const float* W1  = (const float*)d_in[1];
  const float* b1  = (const float*)d_in[2];
  const float* W2  = (const float*)d_in[3];
  const float* b2  = (const float*)d_in[4];
  const float* W3  = (const float*)d_in[5];
  const float* b3  = (const float*)d_in[6];
  // d_in[7] = species (unused: block_index already partitions by species)
  const int*   bidx = (const int*)d_in[8];
  float* out = (float*)d_out;
  unsigned short* ws = (unsigned short*)d_ws;

  const int n1 = NSPEC * NH1 * DIN;
  const int n2 = NSPEC * NH2 * NH1;
  const int n3 = NSPEC * DOUT * NH2;
  const int total = n1 + n2 + n3;                 // 1,310,720 bf16 elems (2.6 MB)
  prep_weights<<<(total + 255) / 256, 256, 0, stream>>>(W1, W2, W3, ws);

  const __bf16* w1t = (const __bf16*)ws;
  const __bf16* w2t = w1t + n1;
  const __bf16* w3t = w2t + n2;

  const size_t smem_bytes = (size_t)2 * TM * STR * sizeof(unsigned short)
                          + (size_t)TM * sizeof(int);  // ~136 KB of the 320 KB WGP LDS
  fused_mlp<<<NSPEC * TILES, 256, smem_bytes, stream>>>(emb, bidx, b1, b2, b3,
                                                        w1t, w2t, w3t, out);
}
